// TemporalPool_89704686944324
// MI455X (gfx1250) — compile-verified, matrix operand order checked by reading, not run on hardware
//
#include <hip/hip_runtime.h>

// ---------------------------------------------------------------------------
// TemporalPool for MI455X (gfx1250, wave32, WMMA).
//   x[64,2048,768] -> LN -> @W_proj[768,256]+b -> h
//   a=tanh(h@W1+b1); w=softmax_T(a.W2); out[64,256]=sum_t h*w
// bf16 WMMA (f32 accum) for both GEMMs.
// x is read once from HBM (stats pass); the GEMM pass re-streams it through
// async global->LDS copies (ASYNCcnt) double-buffered against the WMMAs.
// ---------------------------------------------------------------------------

typedef __attribute__((ext_vector_type(16))) __bf16 v16bf;
typedef __attribute__((ext_vector_type(2)))  __bf16 bf16x2;
typedef __attribute__((ext_vector_type(8)))  float  v8f;

// GCC-style int4 vector with explicit address spaces, matching the builtin
// signature vV4i*1 V4i*3 Ii Ii (global src, LDS dst, imm offset, imm cpol).
typedef int v4i_gcc __attribute__((vector_size(16)));
typedef __attribute__((address_space(1))) v4i_gcc* gv4i_p;
typedef __attribute__((address_space(3))) v4i_gcc* lv4i_p;

#define IN_DIM  768
#define D_MODEL 256
#define T_SEQ   2048
#define B_BATCH 64
#define LN_EPS  1e-5f

#if defined(__has_builtin)
#if __has_builtin(__builtin_amdgcn_global_load_async_to_lds_b128)
#define USE_ASYNC_LDS 1
#endif
#endif

__device__ __forceinline__ void wait_async0() {
#if defined(USE_ASYNC_LDS)
#if __has_builtin(__builtin_amdgcn_s_wait_asynccnt)
  __builtin_amdgcn_s_wait_asynccnt(0);
#else
  asm volatile("s_wait_asynccnt 0" ::: "memory");
#endif
#endif
}

// nan/inf -> 0 (reference nan_to_num). NaN compares false -> 0.
__device__ __forceinline__ float san(float v) {
  return (fabsf(v) <= 3.0e38f) ? v : 0.0f;
}
// native RTE f32 -> bf16 bits (v_cvt, not the 3-op software sequence)
__device__ __forceinline__ unsigned short f2bf(float f) {
  union { __bf16 b; unsigned short u; } c;
  c.b = (__bf16)f;
  return c.u;
}
// packed pair -> one dword
__device__ __forceinline__ unsigned f2bf2(float lo, float hi) {
  union { bf16x2 b; unsigned u; } c;
  c.b = bf16x2{(__bf16)lo, (__bf16)hi};
  return c.u;
}
__device__ __forceinline__ float bf2f(unsigned short h) {
  return __uint_as_float(((unsigned)h) << 16);
}

// ---------------------------------------------------------------------------
// Pack a row-major f32 [K,N] weight matrix into bf16 WMMA B-fragments so each
// lane's 16 bf16 (32B) are contiguous:  P[((kTile*nTiles+nTile)*32+lane)*16+j]
// B-fragment layout mirrors the ISA 16-bit A layout with N in place of M:
//   lane = N%16 + 16*half,  kLocal = (j&7) | (half<<3) | ((j>>3)<<4)
// ---------------------------------------------------------------------------
__global__ void pack_wmma_b(const float* __restrict__ W,
                            unsigned short* __restrict__ P,
                            int K, int N) {
  int pidx = blockIdx.x * 256 + threadIdx.x;
  if (pidx >= K * N) return;
  int j      = pidx & 15;
  int lane   = (pidx >> 4) & 31;
  int frag   = pidx >> 9;
  int nTiles = N >> 4;
  int nTile  = frag % nTiles;
  int kTile  = frag / nTiles;
  int half   = lane >> 4;
  int n      = nTile * 16 + (lane & 15);
  int kLoc   = (j & 7) | (half << 3) | ((j >> 3) << 4);
  int k      = kTile * 32 + kLoc;
  P[pidx] = f2bf(W[(size_t)k * N + n]);
}

// ---------------------------------------------------------------------------
// Main fused kernel: 64 tokens per block, 256 threads (8 waves).
// wave = mSub(4) x nHalf(2):  each wave owns 16 tokens x 128 output channels.
// ---------------------------------------------------------------------------
__launch_bounds__(256)
__global__ void fused_ln_proj_score(
    const float* __restrict__ x,
    const float* __restrict__ gamma,
    const float* __restrict__ beta,
    const float* __restrict__ b_proj,
    const float* __restrict__ b1,
    const float* __restrict__ W2,
    const unsigned short* __restrict__ pWp,  // packed W_proj frags
    const unsigned short* __restrict__ pW1,  // packed W1 frags
    unsigned short* __restrict__ hOut,       // [B*T,256] bf16 bits
    float* __restrict__ logits)              // [B*T]
{
  __shared__ __align__(16) unsigned short aFrag[4][32][16];     // 4 KB
  __shared__ __align__(16) unsigned short hFrag[4][8][32][16];  // 32 KB
#if defined(USE_ASYNC_LDS)
  __shared__ __align__(16) float xStage[2][64][32];             // 16 KB
#endif
  __shared__ float s_mu[64], s_rstd[64], s_logit[64];

  const int tid   = threadIdx.x;
  const int lane  = tid & 31;
  const int wave  = tid >> 5;
  const int mSub  = wave & 3;   // which 16-token subtile
  const int nHalf = wave >> 2;  // which 128-channel half
  const size_t tokenBase = (size_t)blockIdx.x * 64;

#if defined(USE_ASYNC_LDS)
  // kick off async copy of K-chunk 0 (overlaps with the LN stats pass)
  {
#pragma unroll
    for (int q = 0; q < 2; ++q) {
      const int e   = (tid + q * 256) * 4;  // flat f32 index in 64x32 chunk
      const int m   = e >> 5;
      const int col = e & 31;
      const float* src = x + (tokenBase + m) * IN_DIM + col;
      __builtin_amdgcn_global_load_async_to_lds_b128(
          (gv4i_p)src, (lv4i_p)&xStage[0][m][col], 0, 0);
    }
  }
#endif

  // ---- Phase 1: LN statistics (each wave: 8 tokens; sanitized) ----
  for (int tt = 0; tt < 8; ++tt) {
    const int m = wave * 8 + tt;
    const float4* xr =
        reinterpret_cast<const float4*>(x + (tokenBase + m) * IN_DIM);
    float sum = 0.f, sq = 0.f;
#pragma unroll
    for (int i = 0; i < 6; ++i) {        // 6*32 lanes*4 = 768
      float4 v = xr[lane + 32 * i];
      float a0 = san(v.x), a1 = san(v.y), a2 = san(v.z), a3 = san(v.w);
      sum += a0 + a1 + a2 + a3;
      sq  += a0*a0 + a1*a1 + a2*a2 + a3*a3;
    }
#pragma unroll
    for (int off = 16; off; off >>= 1) {
      sum += __shfl_xor(sum, off);
      sq  += __shfl_xor(sq, off);
    }
    if (lane == 0) {
      float mu  = sum * (1.0f / IN_DIM);
      float var = sq * (1.0f / IN_DIM) - mu * mu;
      s_mu[m]   = mu;
      s_rstd[m] = rsqrtf(var + LN_EPS);
    }
  }

  // ---- Phase 2: GEMM1  h = LN(x) @ W_proj   (K=768, 24 chunks of 32) ----
  v8f c[8] = {};
  for (int kc = 0; kc < 24; ++kc) {
    wait_async0();    // chunk kc resident in xStage[kc&1] (async path)
    __syncthreads();  // stats visible (kc==0); prior aFrag consumers done

#if defined(USE_ASYNC_LDS)
    if (kc + 1 < 24) {  // prefetch next chunk into the other buffer
#pragma unroll
      for (int q = 0; q < 2; ++q) {
        const int e   = (tid + q * 256) * 4;
        const int m   = e >> 5;
        const int col = e & 31;
        const float* src = x + (tokenBase + m) * IN_DIM + (kc + 1) * 32 + col;
        __builtin_amdgcn_global_load_async_to_lds_b128(
            (gv4i_p)src, (lv4i_p)&xStage[(kc + 1) & 1][m][col], 0, 0);
      }
    }
#endif

    // producer: normalize this K-chunk into A-fragments (paired dword stores)
#pragma unroll
    for (int i = 0; i < 4; ++i) {
      const int pairIdx = tid + 256 * i;       // 1024 pairs total
      const int m  = pairIdx >> 4;             // token within tile
      const int kl = (pairIdx & 15) * 2;       // even K within chunk
      const int col = kc * 32 + kl;
#if defined(USE_ASYNC_LDS)
      const float2 xv = *reinterpret_cast<const float2*>(&xStage[kc & 1][m][kl]);
#else
      const float2 xv = *reinterpret_cast<const float2*>(
          x + (tokenBase + m) * IN_DIM + col);
#endif
      const float2 gv = reinterpret_cast<const float2*>(gamma)[col >> 1];
      const float2 bv = reinterpret_cast<const float2*>(beta)[col >> 1];
      const float mu = s_mu[m], rs = s_rstd[m];
      const float xn0 = (san(xv.x) - mu) * rs * gv.x + bv.x;
      const float xn1 = (san(xv.y) - mu) * rs * gv.y + bv.y;
      const int dl = (m & 15) + (((kl >> 3) & 1) << 4);
      const int dj = (kl & 7) | ((kl >> 4) << 3);  // even; dj+1 is partner
      *reinterpret_cast<unsigned*>(&aFrag[m >> 4][dl][dj]) = f2bf2(xn0, xn1);
    }
    __syncthreads();

    union { uint4 q[2]; v16bf v; } A;
    A.q[0] = *reinterpret_cast<const uint4*>(&aFrag[mSub][lane][0]);
    A.q[1] = *reinterpret_cast<const uint4*>(&aFrag[mSub][lane][8]);
#pragma unroll
    for (int nt = 0; nt < 8; ++nt) {
      union { uint4 q[2]; v16bf v; } Bf;
      const unsigned short* bp =
          pWp + (((size_t)(kc * 16 + nHalf * 8 + nt) * 32 + lane) << 4);
      Bf.q[0] = reinterpret_cast<const uint4*>(bp)[0];
      Bf.q[1] = reinterpret_cast<const uint4*>(bp)[1];
      c[nt] = __builtin_amdgcn_wmma_f32_16x16x32_bf16(
          false, A.v, false, Bf.v, (short)0, c[nt], false, false);
    }
  }

  // ---- bias, emit h (global bf16 for pooling) + hFrag (LDS, GEMM2 A) ----
  const int rowHalf = lane >> 4;  // C layout: VGPR r -> M=r (lanes<16) / r+8
  const int nl      = lane & 15;
#pragma unroll
  for (int nt = 0; nt < 8; ++nt) {
    const int n    = nHalf * 128 + nt * 16 + nl;
    const float bb = b_proj[n];
    const int k2   = n >> 5;
    const int kl   = n & 31;
    const int dj   = (kl & 7) | ((kl >> 4) << 3);
    const int dlHi = (((kl >> 3) & 1) << 4);
#pragma unroll
    for (int r = 0; r < 8; ++r) {
      const int m = r + 8 * rowHalf;                 // 0..15 in subtile
      unsigned short hb = f2bf(c[nt][r] + bb);
      hFrag[mSub][k2][(m & 15) + dlHi][dj] = hb;
      hOut[(tokenBase + mSub * 16 + m) * D_MODEL + n] = hb;
    }
  }
  if (tid < 64) s_logit[tid] = 0.0f;
  __syncthreads();

  // ---- GEMM2  a = tanh(h @ W1 + b1)   (K=256, 8 chunks) ----
  v8f c2[8] = {};
  for (int kc = 0; kc < 8; ++kc) {
    union { uint4 q[2]; v16bf v; } A;
    A.q[0] = *reinterpret_cast<const uint4*>(&hFrag[mSub][kc][lane][0]);
    A.q[1] = *reinterpret_cast<const uint4*>(&hFrag[mSub][kc][lane][8]);
#pragma unroll
    for (int nt = 0; nt < 8; ++nt) {
      union { uint4 q[2]; v16bf v; } Bf;
      const unsigned short* bp =
          pW1 + (((size_t)(kc * 16 + nHalf * 8 + nt) * 32 + lane) << 4);
      Bf.q[0] = reinterpret_cast<const uint4*>(bp)[0];
      Bf.q[1] = reinterpret_cast<const uint4*>(bp)[1];
      c2[nt] = __builtin_amdgcn_wmma_f32_16x16x32_bf16(
          false, A.v, false, Bf.v, (short)0, c2[nt], false, false);
    }
  }

  // ---- logits: tanh then dot with W2, reduce across the 16 N-lanes ----
  float part[8];
#pragma unroll
  for (int r = 0; r < 8; ++r) part[r] = 0.0f;
#pragma unroll
  for (int nt = 0; nt < 8; ++nt) {
    const int n     = nHalf * 128 + nt * 16 + nl;
    const float bb  = b1[n];
    const float w2v = W2[n];
#pragma unroll
    for (int r = 0; r < 8; ++r)
      part[r] += tanhf(c2[nt][r] + bb) * w2v;
  }
#pragma unroll
  for (int off = 1; off < 16; off <<= 1) {
#pragma unroll
    for (int r = 0; r < 8; ++r) part[r] += __shfl_xor(part[r], off);
  }
  if (nl == 0) {
#pragma unroll
    for (int r = 0; r < 8; ++r)  // two nHalf waves accumulate per token
      atomicAdd(&s_logit[mSub * 16 + r + 8 * rowHalf], part[r]);
  }
  __syncthreads();
  if (tid < 64) logits[tokenBase + tid] = s_logit[tid];
  // (b2 omitted: softmax is shift-invariant, it cancels exactly)
}

// ---------------------------------------------------------------------------
// Per-batch softmax over T then weighted pooling of h. One block per b.
// ---------------------------------------------------------------------------
__launch_bounds__(256)
__global__ void softmax_pool(const float* __restrict__ logits,
                             const unsigned short* __restrict__ h,
                             float* __restrict__ out) {
  __shared__ float sw[T_SEQ];  // exp(l - max)
  __shared__ float red[8];
  const int b = blockIdx.x, tid = threadIdx.x;
  const int lane = tid & 31, wave = tid >> 5;
  const float* lrow = logits + (size_t)b * T_SEQ;

  float mx = -3.4e38f;
  for (int t = tid; t < T_SEQ; t += 256) mx = fmaxf(mx, lrow[t]);
#pragma unroll
  for (int off = 16; off; off >>= 1) mx = fmaxf(mx, __shfl_xor(mx, off));
  if (lane == 0) red[wave] = mx;
  __syncthreads();
  mx = red[0];
#pragma unroll
  for (int i = 1; i < 8; ++i) mx = fmaxf(mx, red[i]);
  __syncthreads();

  float sum = 0.0f;
  for (int t = tid; t < T_SEQ; t += 256) {
    float e = expf(lrow[t] - mx);
    sw[t] = e;
    sum += e;
  }
#pragma unroll
  for (int off = 16; off; off >>= 1) sum += __shfl_xor(sum, off);
  if (lane == 0) red[wave] = sum;
  __syncthreads();
  sum = 0.0f;
#pragma unroll
  for (int i = 0; i < 8; ++i) sum += red[i];
  const float inv = 1.0f / sum;

  // pooling: thread tid owns channel d=tid; coalesced bf16 h reads (L2-hot)
  const unsigned short* hb = h + (size_t)b * T_SEQ * D_MODEL;
  float acc = 0.0f;
  for (int t = 0; t < T_SEQ; t += 4) {
#pragma unroll
    for (int u = 0; u < 4; ++u)
      acc += sw[t + u] * bf2f(hb[(size_t)(t + u) * D_MODEL + tid]);
  }
  out[b * D_MODEL + tid] = acc * inv;
}

// ---------------------------------------------------------------------------
extern "C" void kernel_launch(void* const* d_in, const int* in_sizes, int n_in,
                              void* d_out, int out_size, void* d_ws,
                              size_t ws_size, hipStream_t stream) {
  (void)in_sizes; (void)n_in; (void)out_size; (void)ws_size;
  const float* x     = (const float*)d_in[0];
  const float* gamma = (const float*)d_in[1];
  const float* beta  = (const float*)d_in[2];
  const float* Wp    = (const float*)d_in[3];
  const float* bp    = (const float*)d_in[4];
  const float* W1    = (const float*)d_in[5];
  const float* b1    = (const float*)d_in[6];
  const float* W2    = (const float*)d_in[7];
  // d_in[8] = b2 : cancels in softmax
  float* out = (float*)d_out;

  // workspace layout (needs ~68 MB):
  //   [0)          packed W_proj bf16  768*256*2 = 393216 B
  //   [393216)     packed W1    bf16   256*256*2 = 131072 B
  //   [524288)     logits f32          131072*4  = 524288 B
  //   [1048576)    h bf16              131072*256*2 = 67108864 B
  char* ws = (char*)d_ws;
  unsigned short* pWp    = (unsigned short*)(ws + 0);
  unsigned short* pW1    = (unsigned short*)(ws + 393216);
  float*          logits = (float*)(ws + 524288);
  unsigned short* hBuf   = (unsigned short*)(ws + 1048576);

  pack_wmma_b<<<(IN_DIM * D_MODEL) / 256, 256, 0, stream>>>(Wp, pWp, IN_DIM,
                                                            D_MODEL);
  pack_wmma_b<<<(D_MODEL * D_MODEL) / 256, 256, 0, stream>>>(W1, pW1, D_MODEL,
                                                             D_MODEL);
  fused_ln_proj_score<<<(B_BATCH * T_SEQ) / 64, 256, 0, stream>>>(
      x, gamma, beta, bp, b1, W2, pWp, pW1, hBuf, logits);
  softmax_pool<<<B_BATCH, 256, 0, stream>>>(logits, hBuf, out);
}
